// Encoder_62886911148217
// MI455X (gfx1250) — matrix operand order repaired and checked by reading, not compiled, not run on hardware
//
#include <hip/hip_runtime.h>
#include <hip/hip_bf16.h>
#include <cstdint>

typedef __bf16 v16bf __attribute__((ext_vector_type(16)));
typedef __bf16 v8bf  __attribute__((ext_vector_type(8)));
typedef __bf16 v4bf  __attribute__((ext_vector_type(4)));
typedef __bf16 v2bf  __attribute__((ext_vector_type(2)));
typedef float  v8f   __attribute__((ext_vector_type(8)));

union FragU { v16bf v; v8bf h[2]; };

#define TM 64
#define TN 128
#define TK 32
#define APAD 40   // bf16 elems per LDS row (32 data + 8 pad) = 80 bytes, 16B aligned

// ---------------------------------------------------------------------------
// fp32 -> bf16 weight conversion (pairs)
// ---------------------------------------------------------------------------
__global__ __launch_bounds__(256)
void cvt_bf16_kernel(const float* __restrict__ src, __bf16* __restrict__ dst, int npair)
{
    int i = blockIdx.x * 256 + threadIdx.x;
    if (i < npair) {
        float2 f = ((const float2*)src)[i];
        v2bf p; p[0] = (__bf16)f.x; p[1] = (__bf16)f.y;
        *(v2bf*)&dst[2 * i] = p;
    }
}

// ---------------------------------------------------------------------------
// Issue one 128x32 bf16 W tile as async global->LDS copies (2 x b128/thread)
// ---------------------------------------------------------------------------
__device__ __forceinline__
void issue_w_tile(const __bf16* __restrict__ W, __bf16* dstLds,
                  int n0, int N_, int K_, int kk, int tid)
{
#pragma unroll
    for (int i = 0; i < 2; ++i) {
        int q = i * 256 + tid;          // chunk 0..511
        int r = q >> 2, c = q & 3;      // row 0..127, 16B chunk 0..3
        int gn = n0 + r;
        if (gn < N_) {
            const __bf16* gp = &W[(size_t)gn * K_ + kk + c * 8];
            unsigned lo = (unsigned)(uintptr_t)&dstLds[r * APAD + c * 8];
            asm volatile("global_load_async_to_lds_b128 %0, %1, off"
                         :: "v"(lo), "v"(gp) : "memory");
        }
    }
}

// ---------------------------------------------------------------------------
// GEMM: C[4096,N] = act(A[4096,K](lda,fp32) @ Wbf[N,K]^T + bias), bf16 WMMA
// A tile: fp32->bf16 on LDS store; W tile: double-buffered async DMA to LDS
// act: 0 none, 1 softplus, 2 sigmoid
// ---------------------------------------------------------------------------
__global__ __launch_bounds__(256)
void gemm_bf16_wmma(const float* __restrict__ A, int lda,
                    const __bf16* __restrict__ W,   // [N,K] bf16 row-major
                    const float* __restrict__ bias, // [N] or nullptr
                    float* __restrict__ C, int ldc,
                    int N_, int K_, int act)
{
    __shared__ __align__(16) __bf16 Asb[TM * APAD];
    __shared__ __align__(16) __bf16 Wsb[2][TN * APAD];

    const int tid  = threadIdx.x;
    const int lane = tid & 31;
    const int wave = tid >> 5;     // 8 waves: 4(M) x 2(N)
    const int wm   = wave >> 1;
    const int wn   = wave & 1;
    const int half = lane >> 4;
    const int lrow = lane & 15;
    const int m0 = blockIdx.x * TM;
    const int n0 = blockIdx.y * TN;

    v8f acc[4];
#pragma unroll
    for (int s = 0; s < 4; ++s)
#pragma unroll
        for (int r = 0; r < 8; ++r) acc[s][r] = 0.f;

    // Prime the async pipeline with the first W tile
    issue_w_tile(W, Wsb[0], n0, N_, K_, 0, tid);

    int p = 0;
    for (int kk = 0; kk < K_; kk += TK) {
        // --- A tile 64x32: 2 float4 loads in flight, then cvt + b64 LDS store
        float4 fa[2];
#pragma unroll
        for (int i = 0; i < 2; ++i) {
            int q = i * 256 + tid;         // float4 chunk 0..511
            int r = q >> 3, c4 = q & 7;    // row 0..63, chunk 0..7
            fa[i] = *(const float4*)&A[(size_t)(m0 + r) * lda + kk + 4 * c4];
        }
#pragma unroll
        for (int i = 0; i < 2; ++i) {
            int q = i * 256 + tid;
            int r = q >> 3, c4 = q & 7;
            v4bf pk;
            pk[0] = (__bf16)fa[i].x; pk[1] = (__bf16)fa[i].y;
            pk[2] = (__bf16)fa[i].z; pk[3] = (__bf16)fa[i].w;
            *(v4bf*)&Asb[r * APAD + 4 * c4] = pk;   // 8B-aligned ds_store_b64
        }
        // Current W tile must have landed; publish A + W to all waves
        asm volatile("s_wait_asynccnt 0x0" ::: "memory");
        __syncthreads();

        // Overlap: kick off next W tile into the other buffer
        if (kk + TK < K_) {
            issue_w_tile(W, Wsb[p ^ 1], n0, N_, K_, kk + TK, tid);
            __builtin_prefetch(&A[(size_t)(m0 + (tid & 63)) * lda + kk + TK], 0, 1);
        }

        // --- A fragment: two contiguous 16B chunks per lane ---
        FragU af;
        {
            const __bf16* Ar = &Asb[(wm * 16 + lrow) * APAD + 8 * half];
            af.h[0] = *(const v8bf*)(Ar);
            af.h[1] = *(const v8bf*)(Ar + 16);
        }
#pragma unroll
        for (int s = 0; s < 4; ++s) {
            FragU bfg;
            const __bf16* Wr = &Wsb[p][(wn * 64 + s * 16 + lrow) * APAD + 8 * half];
            bfg.h[0] = *(const v8bf*)(Wr);
            bfg.h[1] = *(const v8bf*)(Wr + 16);
            acc[s] = __builtin_amdgcn_wmma_f32_16x16x32_bf16(
                false, af.v, false, bfg.v, (short)0, acc[s], false, false);
        }
        __syncthreads();   // protect Asb / Wsb[p^1] reuse
        p ^= 1;
    }

    // --- Epilogue ---
#pragma unroll
    for (int s = 0; s < 4; ++s) {
        int col = n0 + wn * 64 + s * 16 + lrow;
        if (col >= N_) continue;
        float bv = bias ? bias[col] : 0.f;
#pragma unroll
        for (int r = 0; r < 8; ++r) {
            int row = m0 + wm * 16 + half * 8 + r;
            float v = acc[s][r] + bv;
            if (act == 1)      v = (v > 20.f) ? v : log1pf(__expf(v));
            else if (act == 2) v = 1.f / (1.f + __expf(-v));
            C[(size_t)row * ldc + col] = v;
        }
    }
}

// ---------------------------------------------------------------------------
// Causal depthwise conv (width 1 or 4) + SiLU.  xz:[M,2048] -> xi:[M,1024]
// ---------------------------------------------------------------------------
__global__ __launch_bounds__(256)
void conv_silu_kernel(const float* __restrict__ xz,
                      const float* __restrict__ cw,  // [1024, dconv]
                      const float* __restrict__ cb,  // [1024]
                      float* __restrict__ xi, int dconv)
{
    int idx = blockIdx.x * 256 + threadIdx.x;   // M*1024
    int m = idx >> 10, d = idx & 1023;
    int t = m & 511;
    float acc = cb[d];
    for (int k = 0; k < dconv; ++k) {
        int st = t - (dconv - 1) + k;
        if (st >= 0)
            acc += cw[d * dconv + k] * xz[(size_t)(m - (dconv - 1 - k)) * 2048 + d];
    }
    xi[idx] = acc / (1.f + __expf(-acc));
}

// ---------------------------------------------------------------------------
// Selective scan: one thread per (batch, channel).  dir=+1 fwd, -1 bwd.
// Fused epilogue: y = (scan + xi*D) * silu(z)
// ---------------------------------------------------------------------------
__global__ __launch_bounds__(256)
void scan_kernel(const float* __restrict__ dt,   // [M,1024]
                 const float* __restrict__ xi,   // [M,1024]
                 const float* __restrict__ dbl,  // [M,64]  (B at 32.., C at 48..)
                 const float* __restrict__ xz,   // [M,2048] (z at +1024)
                 const float* __restrict__ A_log,// [1024,16]
                 const float* __restrict__ Dp,   // [1024]
                 float* __restrict__ yact,       // [M,1024]
                 int dir)
{
    __shared__ float sBC[32];
    int d = blockIdx.x * 256 + threadIdx.x;
    int b = blockIdx.y;

    float Avec[16], h[16];
#pragma unroll
    for (int s = 0; s < 16; ++s) { Avec[s] = -__expf(A_log[d * 16 + s]); h[s] = 0.f; }
    float Dv = Dp[d];

    for (int step = 0; step < 512; ++step) {
        int t = (dir > 0) ? step : (511 - step);
        size_t row = (size_t)b * 512 + t;
        __syncthreads();
        if (threadIdx.x < 32) sBC[threadIdx.x] = dbl[row * 64 + 32 + threadIdx.x];
        __syncthreads();
        float dtv = dt[row * 1024 + d];
        float xv  = xi[row * 1024 + d];
        float dx  = dtv * xv;
        float acc = 0.f;
#pragma unroll
        for (int s = 0; s < 16; ++s) {
            h[s] = __expf(dtv * Avec[s]) * h[s] + dx * sBC[s];
            acc += h[s] * sBC[16 + s];
        }
        float zv = xz[row * 2048 + 1024 + d];
        yact[row * 1024 + d] = (acc + xv * Dv) * (zv / (1.f + __expf(-zv)));
    }
}

// ---------------------------------------------------------------------------
// Gate mix: ym = zg*y_f + (1-zg)*y_b
// ---------------------------------------------------------------------------
__global__ __launch_bounds__(256)
void mix_kernel(const float* __restrict__ zg, const float* __restrict__ ycat,
                float* __restrict__ ym)
{
    int idx = blockIdx.x * 256 + threadIdx.x;   // M*512
    int m = idx >> 9, c = idx & 511;
    float g  = zg[idx];
    float yf = ycat[(size_t)m * 1024 + c];
    float yb = ycat[(size_t)m * 1024 + 512 + c];
    ym[idx] = g * yf + (1.f - g) * yb;
}

// ---------------------------------------------------------------------------
// Fused residual + LayerNorm over D=512:  dst = LN(a*(mul?)+res)
// ---------------------------------------------------------------------------
__global__ __launch_bounds__(256)
void ln_kernel(const float* __restrict__ a, const float* __restrict__ mul,
               const float* __restrict__ res,
               const float* __restrict__ g, const float* __restrict__ beta,
               float* __restrict__ dst)
{
    __shared__ float red[256];
    int row = blockIdx.x, tid = threadIdx.x;
    size_t base = (size_t)row * 512;
    float x0 = a[base + tid]       * (mul ? mul[base + tid]       : 1.f) + res[base + tid];
    float x1 = a[base + 256 + tid] * (mul ? mul[base + 256 + tid] : 1.f) + res[base + 256 + tid];

    red[tid] = x0 + x1;
    __syncthreads();
    for (int s = 128; s > 0; s >>= 1) { if (tid < s) red[tid] += red[tid + s]; __syncthreads(); }
    float mu = red[0] * (1.f / 512.f);
    __syncthreads();
    float d0 = x0 - mu, d1 = x1 - mu;
    red[tid] = d0 * d0 + d1 * d1;
    __syncthreads();
    for (int s = 128; s > 0; s >>= 1) { if (tid < s) red[tid] += red[tid + s]; __syncthreads(); }
    float rstd = rsqrtf(red[0] * (1.f / 512.f) + 1e-5f);
    dst[base + tid]       = d0 * rstd * g[tid]       + beta[tid];
    dst[base + 256 + tid] = d1 * rstd * g[tid + 256] + beta[tid + 256];
}

// ---------------------------------------------------------------------------
// 512x512 per-batch tiled transpose
// ---------------------------------------------------------------------------
__global__ void transpose512(const float* __restrict__ src, float* __restrict__ dst)
{
    __shared__ float tile[32][33];
    int b = blockIdx.z;
    int x0 = blockIdx.x * 32, y0 = blockIdx.y * 32;
    const float* s = src + (size_t)b * 512 * 512;
    float*       d = dst + (size_t)b * 512 * 512;
#pragma unroll
    for (int i = 0; i < 4; ++i)
        tile[threadIdx.y + i * 8][threadIdx.x] =
            s[(size_t)(y0 + threadIdx.y + i * 8) * 512 + x0 + threadIdx.x];
    __syncthreads();
#pragma unroll
    for (int i = 0; i < 4; ++i)
        d[(size_t)(x0 + threadIdx.y + i * 8) * 512 + y0 + threadIdx.x] =
            tile[threadIdx.x][threadIdx.y + i * 8];
}

// ---------------------------------------------------------------------------
// Host-side orchestration
// ---------------------------------------------------------------------------
struct MP { const float *A_log,*D,*conv_b,*conv_w,*dt_b,*dt_w,*in_w,*out_w,*x_w; };
struct MPB { __bf16 *in_w,*x_w,*dt_w,*out_w; };

static MP get_mp(void* const* d_in, int base) {
    MP p;
    p.A_log  = (const float*)d_in[base + 0];
    p.D      = (const float*)d_in[base + 1];
    p.conv_b = (const float*)d_in[base + 2];
    p.conv_w = (const float*)d_in[base + 3];
    p.dt_b   = (const float*)d_in[base + 4];
    p.dt_w   = (const float*)d_in[base + 5];
    p.in_w   = (const float*)d_in[base + 6];
    p.out_w  = (const float*)d_in[base + 7];
    p.x_w    = (const float*)d_in[base + 8];
    return p;
}

static void cvt_w(const float* src, __bf16* dst, int n, hipStream_t stream) {
    int npair = n / 2;
    cvt_bf16_kernel<<<(npair + 255) / 256, 256, 0, stream>>>(src, dst, npair);
}

static void run_mamba(const float* in, const MP& p, const MPB& pb16, int dconv, int dir,
                      float* Cout, int ldc,
                      float* XZ, float* XI, float* DBL, float* DT, float* YA,
                      hipStream_t stream)
{
    const int M = 4096;
    gemm_bf16_wmma<<<dim3(M / 64, 2048 / 128), 256, 0, stream>>>(
        in, 512, pb16.in_w, nullptr, XZ, 2048, 2048, 512, 0);
    conv_silu_kernel<<<M * 1024 / 256, 256, 0, stream>>>(XZ, p.conv_w, p.conv_b, XI, dconv);
    gemm_bf16_wmma<<<dim3(M / 64, 1), 256, 0, stream>>>(
        XI, 1024, pb16.x_w, nullptr, DBL, 64, 64, 1024, 0);
    gemm_bf16_wmma<<<dim3(M / 64, 1024 / 128), 256, 0, stream>>>(
        DBL, 64, pb16.dt_w, p.dt_b, DT, 1024, 1024, 32, 1);
    scan_kernel<<<dim3(4, 8), 256, 0, stream>>>(DT, XI, DBL, XZ, p.A_log, p.D, YA, dir);
    gemm_bf16_wmma<<<dim3(M / 64, 512 / 128), 256, 0, stream>>>(
        YA, 1024, pb16.out_w, nullptr, Cout, ldc, 512, 1024, 0);
}

extern "C" void kernel_launch(void* const* d_in, const int* in_sizes, int n_in,
                              void* d_out, int out_size, void* d_ws, size_t ws_size,
                              hipStream_t stream)
{
    (void)in_sizes; (void)n_in; (void)out_size; (void)ws_size;
    const int M = 4096;

    const float* x = (const float*)d_in[0];
    MP pf = get_mp(d_in, 1);
    MP pb = get_mp(d_in, 10);
    MP pt = get_mp(d_in, 19);
    const float* gate_w = (const float*)d_in[28];
    const float* gate_b = (const float*)d_in[29];
    const float* cm_w   = (const float*)d_in[30];
    const float* cm_b   = (const float*)d_in[31];
    const float* ln_g   = (const float*)d_in[32];
    const float* ln_b   = (const float*)d_in[33];
    float* out = (float*)d_out;

    // ---- fp32 workspace ----
    float* w = (float*)d_ws;
    float* XZ   = w;                       // [M,2048]
    float* XI   = XZ   + (size_t)M * 2048; // [M,1024]
    float* DBL  = XI   + (size_t)M * 1024; // [M,64]
    float* DT   = DBL  + (size_t)M * 64;   // [M,1024]
    float* YA   = DT   + (size_t)M * 1024; // [M,1024]
    float* YCAT = YA   + (size_t)M * 1024; // [M,1024]
    float* ZG   = YCAT + (size_t)M * 1024; // [M,512]
    float* TMP  = ZG   + (size_t)M * 512;  // [M,512]
    float* Y0   = TMP  + (size_t)M * 512;  // [M,512]
    float* Y1   = Y0   + (size_t)M * 512;  // [M,512]

    // ---- bf16 weight workspace (converted once per launch) ----
    __bf16* bw = (__bf16*)(Y1 + (size_t)M * 512);
    MPB b16[3];
    const MP* ps[3] = { &pf, &pb, &pt };
    for (int i = 0; i < 3; ++i) {
        b16[i].in_w  = bw; bw += 2048 * 512;
        b16[i].x_w   = bw; bw += 64 * 1024;
        b16[i].dt_w  = bw; bw += 1024 * 32;
        b16[i].out_w = bw; bw += 512 * 1024;
        cvt_w(ps[i]->in_w,  b16[i].in_w,  2048 * 512, stream);
        cvt_w(ps[i]->x_w,   b16[i].x_w,   64 * 1024,  stream);
        cvt_w(ps[i]->dt_w,  b16[i].dt_w,  1024 * 32,  stream);
        cvt_w(ps[i]->out_w, b16[i].out_w, 512 * 1024, stream);
    }
    __bf16* gate_wb = bw; bw += 512 * 1024;
    __bf16* cm_wb   = bw; bw += 512 * 512;
    cvt_w(gate_w, gate_wb, 512 * 1024, stream);
    cvt_w(cm_w,   cm_wb,   512 * 512,  stream);

    // ---- pipeline ----
    run_mamba(x, pf, b16[0], 1, +1, YCAT,       1024, XZ, XI, DBL, DT, YA, stream);
    run_mamba(x, pb, b16[1], 1, -1, YCAT + 512, 1024, XZ, XI, DBL, DT, YA, stream);

    gemm_bf16_wmma<<<dim3(M / 64, 512 / 128), 256, 0, stream>>>(
        YCAT, 1024, gate_wb, gate_b, ZG, 512, 512, 1024, 2);
    mix_kernel<<<M * 512 / 256, 256, 0, stream>>>(ZG, YCAT, TMP);
    gemm_bf16_wmma<<<dim3(M / 64, 512 / 128), 256, 0, stream>>>(
        TMP, 512, cm_wb, cm_b, Y1, 512, 512, 512, 0);
    ln_kernel<<<M, 256, 0, stream>>>(Y1, nullptr, x, ln_g, ln_b, Y0);

    transpose512<<<dim3(16, 16, 8), dim3(32, 8), 0, stream>>>(Y0, TMP);
    run_mamba(TMP, pt, b16[2], 4, +1, ZG, 512, XZ, XI, DBL, DT, YA, stream);
    transpose512<<<dim3(16, 16, 8), dim3(32, 8), 0, stream>>>(ZG, Y1);

    ln_kernel<<<M, 256, 0, stream>>>(Y1, Y0, x, ln_g, ln_b, out);
}
